// CodeBook_44220983279678
// MI455X (gfx1250) — matrix operand order repaired and checked by reading, not compiled, not run on hardware
//
#include <hip/hip_runtime.h>
#include <hip/hip_bf16.h>
#include <stdint.h>

typedef __attribute__((ext_vector_type(16))) __bf16 bf16x16;
typedef __attribute__((ext_vector_type(8)))  __bf16 bf16x8;
typedef __attribute__((ext_vector_type(8)))  float  f32x8;
typedef __attribute__((ext_vector_type(4)))  float  f32x4;

#define NUM_CODES 1024
#define CDIM      256
#define NROWS     (32 * 64 * 64)         // 131072 spatial vectors
#define HW        4096                    // 64*64
#define XQ_ELEMS  ((size_t)NROWS * CDIM)  // 33554432
#define ASTRIDE   264                     // A LDS row stride (halves): 528B, 16B-aligned, bank offset 4
#define BSTRIDE   264                     // B LDS row stride (halves)

union ABFrag { bf16x16 v; bf16x8 h[2]; };

// ---------------------------------------------------------------------------
// Kernel 1: emb (1024x256 f32) -> bf16 hi/lo split + ||e||^2 per code.
// ---------------------------------------------------------------------------
__global__ void prep_emb(const float* __restrict__ emb,
                         __bf16* __restrict__ eh, __bf16* __restrict__ el,
                         float* __restrict__ enorm) {
    const int n = blockIdx.x;
    const int c = threadIdx.x;
    const float v = emb[n * CDIM + c];
    const __bf16 h = (__bf16)v;
    eh[n * CDIM + c] = h;
    el[n * CDIM + c] = (__bf16)(v - (float)h);
    __shared__ float red[256];
    red[c] = v * v;
    __syncthreads();
    for (int s = 128; s > 0; s >>= 1) {
        if (c < s) red[c] += red[c + s];
        __syncthreads();
    }
    if (c == 0) enorm[n] = red[0];
}

// ---------------------------------------------------------------------------
// Kernel 2: argmin over codes via split-bf16 WMMA distance GEMM.
// Block = 128 threads (4 waves); each wave owns 32 rows (2 M-tiles) of x.
// B tile (16 codes, hi+lo) staged into LDS per block via async global->LDS,
// shared by all 4 waves -> 4x less codebook traffic than streaming per wave.
// argmin_n(||x||^2 + ||e_n||^2 - 2 x.e_n) == argmin_n(||e_n||^2 - 2 x.e_n).
// ---------------------------------------------------------------------------
__global__ __launch_bounds__(128, 1) void argmin_kernel(
    const float* __restrict__ x, const __bf16* __restrict__ eh,
    const __bf16* __restrict__ el, const float* __restrict__ enorm,
    int* __restrict__ idx_out) {
    // A: 4 waves x 32 rows, padded stride. hi then lo.
    __shared__ __bf16 shAHi[4 * 32 * ASTRIDE];   // 67,584 B
    __shared__ __bf16 shALo[4 * 32 * ASTRIDE];   // 67,584 B
    // B staging: 16 codes x 256 ch, hi then lo, padded stride.
    __shared__ __bf16 shB[2 * 16 * BSTRIDE];     // 16,896 B

    const int tid  = threadIdx.x;
    const int lane = tid & 31;
    const int wave = tid >> 5;
    const int rowbase = blockIdx.x * 128 + wave * 32;

    __bf16* aHi = shAHi + wave * 32 * ASTRIDE;
    __bf16* aLo = shALo + wave * 32 * ASTRIDE;

    // ---- stage this wave's 32 x-rows (BHWC gather from BCHW), split hi/lo ----
    const int r = lane & 15;
    const float* xrow0;
    const float* xrow1;
    {
        const int n0 = rowbase + r;
        const int n1 = rowbase + 16 + r;
        xrow0 = x + (size_t)(n0 >> 12) * (CDIM * HW) + (n0 & 4095);
        xrow1 = x + (size_t)(n1 >> 12) * (CDIM * HW) + (n1 & 4095);
    }
    #pragma unroll 4
    for (int it = 0; it < 128; ++it) {
        const int c = it * 2 + (lane >> 4);       // lanes 0-15 even c, 16-31 odd c
        const float v0 = xrow0[(size_t)c * HW];   // coalesced: rows contiguous in hw
        const float v1 = xrow1[(size_t)c * HW];
        const __bf16 h0 = (__bf16)v0;
        const __bf16 h1 = (__bf16)v1;
        aHi[r * ASTRIDE + c]        = h0;
        aLo[r * ASTRIDE + c]        = (__bf16)(v0 - (float)h0);
        aHi[(16 + r) * ASTRIDE + c] = h1;
        aLo[(16 + r) * ASTRIDE + c] = (__bf16)(v1 - (float)h1);
    }

    // ---- running per-lane argmin state, two M-tiles ----
    float best0[8], best1[8];
    int   bestn0[8], bestn1[8];
    #pragma unroll
    for (int i = 0; i < 8; ++i) {
        best0[i] = 3.4e38f; best1[i] = 3.4e38f; bestn0[i] = 0; bestn1[i] = 0;
    }

    const int koffA = (lane < 16) ? 0 : 8;        // A half-wave k interleave
    const int koffB = (lane < 16) ? 0 : 8;        // B half-wave k interleave
    const __bf16* aHiR0 = aHi + r * ASTRIDE;
    const __bf16* aLoR0 = aLo + r * ASTRIDE;
    const __bf16* aHiR1 = aHi + (16 + r) * ASTRIDE;
    const __bf16* aLoR1 = aLo + (16 + r) * ASTRIDE;
    const __bf16* bHiRow = shB + (lane & 15) * BSTRIDE;            // this lane's code row
    const __bf16* bLoRow = shB + (16 + (lane & 15)) * BSTRIDE;

    // async staging of B tile 'tile' into LDS (16KB, 1024 x 16B chunks, 128 thr x 8)
    const uint64_t ehB = (uint64_t)(uintptr_t)eh;
    const uint64_t elB = (uint64_t)(uintptr_t)el;
    const uint32_t bLds = (uint32_t)(uintptr_t)(void*)shB;

    auto stageB = [&](int tile) {
        #pragma unroll
        for (int i = 0; i < 8; ++i) {
            const int chunk = i * 128 + tid;       // 0..1023
            const int arr   = chunk >> 9;          // 0=hi, 1=lo
            const int code  = (chunk >> 5) & 15;
            const int k16   = chunk & 31;          // 16B chunk within 512B row
            const uint64_t src = (arr ? elB : ehB)
                               + (uint64_t)tile * (16 * CDIM * 2)
                               + (uint64_t)code * (CDIM * 2) + (uint64_t)k16 * 16;
            const uint32_t dst = bLds + (uint32_t)((arr * 16 + code) * (BSTRIDE * 2) + k16 * 16);
            asm volatile("global_load_async_to_lds_b128 %0, %1, off"
                         :: "v"(dst), "v"(src) : "memory");
        }
        asm volatile("s_wait_asynccnt 0x0" ::: "memory");
    };

    stageB(0);
    __syncthreads();   // A staged (own wave) + B tile 0 visible to all waves

    for (int tile = 0; tile < 64; ++tile) {
        const int ncode = tile * 16 + (lane & 15);
        f32x8 acc0 = {};
        f32x8 acc1 = {};
        #pragma unroll
        for (int kc = 0; kc < 8; ++kc) {           // K = 256 in chunks of 32
            ABFrag bh, bl, a0h, a0l, a1h, a1l;
            const __bf16* bp = bHiRow + kc * 32 + koffB;
            const __bf16* qp = bLoRow + kc * 32 + koffB;
            bh.h[0] = *(const bf16x8*)(bp);
            bh.h[1] = *(const bf16x8*)(bp + 16);
            bl.h[0] = *(const bf16x8*)(qp);
            bl.h[1] = *(const bf16x8*)(qp + 16);
            const __bf16* p0h = aHiR0 + kc * 32 + koffA;
            const __bf16* p0l = aLoR0 + kc * 32 + koffA;
            const __bf16* p1h = aHiR1 + kc * 32 + koffA;
            const __bf16* p1l = aLoR1 + kc * 32 + koffA;
            a0h.h[0] = *(const bf16x8*)(p0h); a0h.h[1] = *(const bf16x8*)(p0h + 16);
            a0l.h[0] = *(const bf16x8*)(p0l); a0l.h[1] = *(const bf16x8*)(p0l + 16);
            a1h.h[0] = *(const bf16x8*)(p1h); a1h.h[1] = *(const bf16x8*)(p1h + 16);
            a1l.h[0] = *(const bf16x8*)(p1l); a1l.h[1] = *(const bf16x8*)(p1l + 16);
            // dot ~= hi*hi + hi*lo + lo*hi  (fp32-class accuracy)
            acc0 = __builtin_amdgcn_wmma_f32_16x16x32_bf16(false, a0h.v, false, bh.v, (short)0, acc0, false, false);
            acc1 = __builtin_amdgcn_wmma_f32_16x16x32_bf16(false, a1h.v, false, bh.v, (short)0, acc1, false, false);
            acc0 = __builtin_amdgcn_wmma_f32_16x16x32_bf16(false, a0h.v, false, bl.v, (short)0, acc0, false, false);
            acc1 = __builtin_amdgcn_wmma_f32_16x16x32_bf16(false, a1h.v, false, bl.v, (short)0, acc1, false, false);
            acc0 = __builtin_amdgcn_wmma_f32_16x16x32_bf16(false, a0l.v, false, bh.v, (short)0, acc0, false, false);
            acc1 = __builtin_amdgcn_wmma_f32_16x16x32_bf16(false, a1l.v, false, bh.v, (short)0, acc1, false, false);
        }
        const float en = enorm[ncode];
        #pragma unroll
        for (int v2 = 0; v2 < 8; ++v2) {
            const float s0 = fmaf(-2.f, acc0[v2], en);
            const float s1 = fmaf(-2.f, acc1[v2], en);
            if (s0 < best0[v2]) { best0[v2] = s0; bestn0[v2] = ncode; }
            if (s1 < best1[v2]) { best1[v2] = s1; bestn1[v2] = ncode; }
        }
        __syncthreads();                       // all waves done reading B tile
        if (tile < 63) {
            stageB(tile + 1);
            __syncthreads();                   // next B tile visible
        }
    }

    // ---- cross-lane argmin (16 lanes per half-wave share each row) ----
    #pragma unroll
    for (int g = 0; g < 2; ++g) {
        #pragma unroll
        for (int v2 = 0; v2 < 8; ++v2) {
            float bv = g ? best1[v2] : best0[v2];
            int   bn = g ? bestn1[v2] : bestn0[v2];
            #pragma unroll
            for (int s = 8; s >= 1; s >>= 1) {
                const float ov = __shfl_xor(bv, s, 32);
                const int   on = __shfl_xor(bn, s, 32);
                if (ov < bv || (ov == bv && on < bn)) { bv = ov; bn = on; }
            }
            if ((lane & 15) == 0) {
                const int m = (lane < 16) ? v2 : v2 + 8;   // C-layout row mapping
                idx_out[rowbase + g * 16 + m] = bn;
            }
        }
    }
}

// ---------------------------------------------------------------------------
// Kernel 3: gather x_q = emb[idx] (BHWC-flat order) and per-row sq-diff vs
// x read at the SAME flat offset (the reference's .view(x.shape) quirk).
// ---------------------------------------------------------------------------
__global__ void quantize_loss(const float* __restrict__ x,
                              const float* __restrict__ emb,
                              const int* __restrict__ idx,
                              float* __restrict__ xq,
                              float* __restrict__ partial) {
    const int n = blockIdx.x;
    const int t = threadIdx.x;                      // 64 threads, 4 floats each
    const int code = idx[n];
    const size_t j = (size_t)n * CDIM + t * 4;
    const f32x4 e  = *(const f32x4*)(emb + (size_t)code * CDIM + t * 4);
    *(f32x4*)(xq + j) = e;
    const f32x4 xv = *(const f32x4*)(x + j);
    float s = 0.f;
    #pragma unroll
    for (int k = 0; k < 4; ++k) { const float d = e[k] - xv[k]; s = fmaf(d, d, s); }
    __shared__ float red[64];
    red[t] = s;
    __syncthreads();
    for (int st = 32; st > 0; st >>= 1) {
        if (t < st) red[t] += red[t + st];
        __syncthreads();
    }
    if (t == 0) partial[n] = red[0];
}

// ---------------------------------------------------------------------------
// Kernel 4: deterministic fixed-order final reduction -> loss scalar.
// loss = mean(d^2) + BETA*mean(d^2) = 1.25 * mean(d^2)
// ---------------------------------------------------------------------------
__global__ void finalize_loss(const float* __restrict__ partial,
                              float* __restrict__ out) {
    __shared__ float red[256];
    const int t = threadIdx.x;
    float s = 0.f;
    for (int i = t; i < NROWS; i += 256) s += partial[i];
    red[t] = s;
    __syncthreads();
    for (int st = 128; st > 0; st >>= 1) {
        if (t < st) red[t] += red[t + st];
        __syncthreads();
    }
    if (t == 0) out[0] = 1.25f * red[0] / (float)XQ_ELEMS;
}

// ---------------------------------------------------------------------------
extern "C" void kernel_launch(void* const* d_in, const int* in_sizes, int n_in,
                              void* d_out, int out_size, void* d_ws, size_t ws_size,
                              hipStream_t stream) {
    const float* x   = (const float*)d_in[0];   // (32,256,64,64) f32
    const float* emb = (const float*)d_in[1];   // (1024,256)     f32

    float* out  = (float*)d_out;
    float* xq   = out;                                   // 33554432 floats
    int*   idx  = (int*)(out + XQ_ELEMS);                // 131072 int32
    float* loss = out + XQ_ELEMS + NROWS;                // 1 float

    char* ws = (char*)d_ws;
    __bf16* eh      = (__bf16*)(ws);                      // 512 KB
    __bf16* el      = (__bf16*)(ws + 512 * 1024);         // 512 KB
    float*  enorm   = (float*) (ws + 1024 * 1024);        // 4 KB
    float*  partial = (float*) (ws + 1024 * 1024 + 4096); // 512 KB

    prep_emb<<<NUM_CODES, 256, 0, stream>>>(emb, eh, el, enorm);
    argmin_kernel<<<NROWS / 128, 128, 0, stream>>>(x, eh, el, enorm, idx);
    quantize_loss<<<NROWS, 64, 0, stream>>>(x, emb, idx, xq, partial);
    finalize_loss<<<1, 256, 0, stream>>>(partial, loss);
}
